// NodeSpecificMLPs_71296457113980
// MI455X (gfx1250) — matrix operand order repaired and checked by reading, not compiled, not run on hardware
//
#include <hip/hip_runtime.h>

// ---------------------------------------------------------------------------
// NodeSpecificMLPs for MI455X (gfx1250): 3-expert MLP with hard routing.
//   layer1: H[16,2048] = relu(X[16,512] @ W1^T + b1)  -> v_wmma_f32_16x16x32_bf16
//   layer2: out[16]    = H @ W2 + b2                  -> fused VALU epilogue
// A (x tile, f32->bf16 RNE) resident in VGPRs; packed bf16 W1 streams into a
// double-buffered LDS tile via GLOBAL_LOAD_ASYNC_TO_LDS_B128 (ASYNCcnt),
// overlapping weight movement with WMMA compute.
// ---------------------------------------------------------------------------

typedef __attribute__((ext_vector_type(16))) __bf16        v16bf;
typedef __attribute__((ext_vector_type(8)))  float         v8f;
typedef __attribute__((ext_vector_type(8)))  unsigned int  v8u;

#define D_IN     512
#define D_HID    2048
#define NTILES   (D_HID / 16)   // 128 n-tiles of 16 hidden units
#define KCHUNKS  (D_IN / 32)    // 16 K-chunks of 32
#define NBLOCKS  (3 * NTILES)   // 384 (expert, n-tile) weight blocks

static __device__ __forceinline__ unsigned bf16rne(unsigned u) {
    // round-to-nearest-even f32 -> bf16 (top 16 bits)
    return (u + 0x7fffu + ((u >> 16) & 1u)) >> 16;
}
static __device__ __forceinline__ unsigned pack2(float lo, float hi) {
    return (bf16rne(__float_as_uint(hi)) << 16) | bf16rne(__float_as_uint(lo));
}

// Async L2->LDS copy of one 16 KB weight block (no VGPR roundtrip, ASYNCcnt).
// Each of the 256 threads moves 64 B with 4x GLOBAL_LOAD_ASYNC_TO_LDS_B128.
static __device__ __forceinline__ void stage_async(const unsigned short* packedW1,
                                                   int blk, unsigned int* dstLds,
                                                   int tid) {
    const char* g = (const char*)packedW1 + (size_t)blk * 16384 + tid * 64;
    // low 32 bits of the LDS-aperture flat address = wave-relative LDS offset
    unsigned l = (unsigned)(uintptr_t)dstLds + (unsigned)(tid * 64);
    #pragma unroll
    for (int i = 0; i < 4; ++i) {
        asm volatile("global_load_async_to_lds_b128 %0, %1, off"
                     :: "v"(l + i * 16), "v"(g + i * 16)
                     : "memory");
    }
}
static __device__ __forceinline__ void wait_async0() {
    asm volatile("s_wait_asynccnt 0x0" ::: "memory");
}

// ---------------------------------------------------------------------------
// Pre-pass: pack 3x W1[2048,512] f32 into bf16 WMMA-B lane layout.
// Element (e, t, c, l, j):  B col N = t*16 + (l%16),  K = c*32 + (l/16)*8 + (j%8) + (j/8)*16
// Packed flat index = (((e*128 + t)*16 + c)*32 + l)*16 + j   (bf16 units)
// ---------------------------------------------------------------------------
__global__ __launch_bounds__(256) void pack_w1_kernel(
    const float* __restrict__ w1c, const float* __restrict__ w1h,
    const float* __restrict__ w1o, unsigned short* __restrict__ packed)
{
    unsigned idx = blockIdx.x * 256u + threadIdx.x;   // 3 * 2^20 total
    unsigned j = idx & 15u;
    unsigned l = (idx >> 4) & 31u;
    unsigned c = (idx >> 9) & 15u;
    unsigned t = (idx >> 13) & 127u;
    unsigned e = idx >> 20;
    const float* W = (e == 0) ? w1c : (e == 1) ? w1h : w1o;
    unsigned n = t * 16u + (l & 15u);
    unsigned k = c * 32u + (l >> 4) * 8u + (j & 7u) + (j >> 3) * 16u;
    packed[idx] = (unsigned short)bf16rne(__float_as_uint(W[n * D_IN + k]));
}

// ---------------------------------------------------------------------------
// Main kernel: 256 threads = 8 wave32; wave w owns rows [blk*128+16w, +16).
// ---------------------------------------------------------------------------
__global__ __launch_bounds__(256) void moe_mlp_kernel(
    const float* __restrict__ x, const int* __restrict__ atomic_nums,
    const unsigned short* __restrict__ packedW1,
    const float* __restrict__ b1c, const float* __restrict__ w2c, const float* __restrict__ b2c,
    const float* __restrict__ b1h, const float* __restrict__ w2h, const float* __restrict__ b2h,
    const float* __restrict__ b1o, const float* __restrict__ w2o, const float* __restrict__ b2o,
    float* __restrict__ out, int nrows)
{
    __shared__ unsigned int ldsB[2][4096];            // 2 x 16 KB double buffer

    const int tid   = threadIdx.x;
    const int lane  = tid & 31;
    const int wave  = tid >> 5;
    const int nIdx  = lane & 15;                      // N column within tile
    const int half  = lane >> 4;                      // lane half (K/M split)
    const int tile0 = blockIdx.x * 128 + wave * 16;   // first row of this wave

    int rowA = tile0 + nIdx;                          // A fragment: M = lane%16
    if (rowA >= nrows) rowA = nrows - 1;              // clamp (stores are guarded)

    // kick off staging of weight block 0 into buffer 0 ASAP
    stage_async(packedW1, 0, &ldsB[0][0], tid);

    // ---- Build resident A operand: 16 x v16bf = full 16x512 bf16 tile ----
    // 16-bit A 16x32 layout: lane half selects K sub-range {h*8..h*8+7, 16+h*8..}
    v16bf a[KCHUNKS];
    const float* xr = x + (size_t)rowA * D_IN;
    #pragma unroll
    for (int kc = 0; kc < KCHUNKS; ++kc) {
        int kb = kc * 32 + half * 8;
        float4 f0 = *(const float4*)(xr + kb);
        float4 f1 = *(const float4*)(xr + kb + 4);
        float4 f2 = *(const float4*)(xr + kb + 16);
        float4 f3 = *(const float4*)(xr + kb + 20);
        v8u u;
        u[0] = pack2(f0.x, f0.y); u[1] = pack2(f0.z, f0.w);
        u[2] = pack2(f1.x, f1.y); u[3] = pack2(f1.z, f1.w);
        u[4] = pack2(f2.x, f2.y); u[5] = pack2(f2.z, f2.w);
        u[6] = pack2(f3.x, f3.y); u[7] = pack2(f3.z, f3.w);
        a[kc] = __builtin_bit_cast(v16bf, u);
    }

    const float* b1s[3] = { b1c, b1h, b1o };
    const float* w2s[3] = { w2c, w2h, w2o };

    float acc[3][8];
    #pragma unroll
    for (int e = 0; e < 3; ++e)
        #pragma unroll
        for (int i = 0; i < 8; ++i) acc[e][i] = 0.0f;

    wait_async0();                                    // block 0 resident in LDS
    __syncthreads();

    #pragma unroll
    for (int e = 0; e < 3; ++e) {
        for (int t = 0; t < NTILES; ++t) {
            const int cur = t & 1;                    // NTILES even -> parity = t&1
            const int nxt = e * NTILES + t + 1;
            if (nxt < NBLOCKS)                        // prefetch next block into
                stage_async(packedW1, nxt, &ldsB[cur ^ 1][0], tid);  // other buffer

            // 16 WMMAs on current buffer; two accumulators break the C RAW chain
            v8f c0 = {0.f, 0.f, 0.f, 0.f, 0.f, 0.f, 0.f, 0.f};
            v8f c1 = {0.f, 0.f, 0.f, 0.f, 0.f, 0.f, 0.f, 0.f};
            const char* bufB = (const char*)&ldsB[cur][0];
            #pragma unroll
            for (int kc = 0; kc < KCHUNKS; kc += 2) {
                v16bf b0 = __builtin_bit_cast(v16bf,
                    *(const v8u*)(bufB + kc * 1024 + lane * 32));
                c0 = __builtin_amdgcn_wmma_f32_16x16x32_bf16(
                        false, a[kc], false, b0, (short)0, c0, false, false);
                v16bf b1f = __builtin_bit_cast(v16bf,
                    *(const v8u*)(bufB + (kc + 1) * 1024 + lane * 32));
                c1 = __builtin_amdgcn_wmma_f32_16x16x32_bf16(
                        false, a[kc + 1], false, b1f, (short)0, c1, false, false);
            }

            // fused layer-2 epilogue: acc += W2[n] * relu(c0 + c1 + b1[n])
            float b1v = b1s[e][t * 16 + nIdx];
            float w2v = w2s[e][t * 16 + nIdx];
            #pragma unroll
            for (int i = 0; i < 8; ++i) {             // C layout: M = i + 8*half
                float hv = c0[i] + c1[i] + b1v;
                hv = hv > 0.0f ? hv : 0.0f;
                acc[e][i] += w2v * hv;
            }

            wait_async0();                            // next buffer fully written
            __syncthreads();                          // visible to all 8 waves
        }
    }

    // ---- reduce acc over the 16 N-lanes of each half-wave ----
    #pragma unroll
    for (int e = 0; e < 3; ++e)
        #pragma unroll
        for (int i = 0; i < 8; ++i) {
            float v = acc[e][i];
            v += __shfl_xor(v, 1, 16);
            v += __shfl_xor(v, 2, 16);
            v += __shfl_xor(v, 4, 16);
            v += __shfl_xor(v, 8, 16);
            acc[e][i] = v;
        }

    float bc = b2c[0], bh = b2h[0], bo = b2o[0];
    if (nIdx == 0) {                                  // lanes 0 and 16 store
        int rbase = tile0 + half * 8;                 // rows M = i + 8*half
        #pragma unroll
        for (int i = 0; i < 8; ++i) {
            int r = rbase + i;
            if (r < nrows) {
                int an = atomic_nums[r];
                float o = (an == 6) ? (acc[0][i] + bc)
                        : (an == 1) ? (acc[1][i] + bh)
                                    : (acc[2][i] + bo);
                out[r] = o;
            }
        }
    }
}

extern "C" void kernel_launch(void* const* d_in, const int* in_sizes, int n_in,
                              void* d_out, int out_size, void* d_ws, size_t ws_size,
                              hipStream_t stream)
{
    const float* x   = (const float*)d_in[0];
    const int*   an  = (const int*)  d_in[1];   // JAX x64 disabled -> int32
    const float* cW1 = (const float*)d_in[2];
    const float* cb1 = (const float*)d_in[3];
    const float* cW2 = (const float*)d_in[4];
    const float* cb2 = (const float*)d_in[5];
    const float* hW1 = (const float*)d_in[6];
    const float* hb1 = (const float*)d_in[7];
    const float* hW2 = (const float*)d_in[8];
    const float* hb2 = (const float*)d_in[9];
    const float* oW1 = (const float*)d_in[10];
    const float* ob1 = (const float*)d_in[11];
    const float* oW2 = (const float*)d_in[12];
    const float* ob2 = (const float*)d_in[13];

    unsigned short* packed = (unsigned short*)d_ws;   // needs 3*2048*512*2 = 6.3 MB
    int nrows = in_sizes[1];

    // pre-pack W1 (f32 -> bf16, WMMA B lane layout): 3*2^20 elements
    pack_w1_kernel<<<(3 * 1048576) / 256, 256, 0, stream>>>(cW1, hW1, oW1, packed);

    int blocks = (nrows + 127) / 128;                 // 128 rows per workgroup
    moe_mlp_kernel<<<blocks, 256, 0, stream>>>(
        x, an, packed,
        cb1, cW2, cb2, hb1, hW2, hb2, ob1, oW2, ob2,
        (float*)d_out, nrows);
}
